// MultiHeadCausalSelfAttention_28724741276219
// MI455X (gfx1250) — matrix operand order repaired and checked by reading, not compiled, not run on hardware
//
#include <hip/hip_runtime.h>
#include <hip/hip_bf16.h>

// MI455X / gfx1250 multi-head causal self-attention.
// All three matmul stages (QKV proj, QK^T / PV, out proj) run on
// v_wmma_f32_16x16x32_bf16 (bf16 in, fp32 accumulate).
// V is stored transposed (B,H,D,T) so every WMMA operand fragment is
// loaded with contiguous global_load_b128.

#define BATCH 4
#define TSEQ  2048
#define CDIM  1024
#define NHEAD 16
#define HDIM  64
#define MROWS (BATCH * TSEQ)   /* 8192 */
#define NQKV  (3 * CDIM)       /* 3072 */

typedef __bf16 bf16_t;
typedef __attribute__((ext_vector_type(16))) __bf16 v16bf;
typedef __attribute__((ext_vector_type(8)))  __bf16 v8bf;
typedef __attribute__((ext_vector_type(8)))  float  v8f;

union BFrag { v16bf v; v8bf h[2]; };

__device__ __forceinline__ v8f wmma_bf16(v16bf a, v16bf b, v8f c) {
  // 8 args: (neg_a, A, neg_b, B, c_mod, C, reuse_a, reuse_b)
  return __builtin_amdgcn_wmma_f32_16x16x32_bf16(false, a, false, b,
                                                 (short)0, c, false, false);
}

// A-matrix fragment (16x32 bf16, row-major source, leading dim `ld`):
// lanes 0-15 hold K{0..7,16..23} of row M=lane, lanes 16-31 hold K{8..15,24..31}.
__device__ __forceinline__ BFrag load_a16x32(const bf16_t* __restrict__ base,
                                             int row0, int ld, int k0, int lane) {
  BFrag f;
  const int kb = k0 + ((lane & 16) ? 8 : 0);
  const bf16_t* p = base + (size_t)(row0 + (lane & 15)) * (size_t)ld + kb;
  f.h[0] = *(const v8bf*)(p);
  f.h[1] = *(const v8bf*)(p + 16);
  return f;
}

// B-matrix fragment (32x16 bf16) from an N-major (transposed) source:
// lane owns column N=col0+(lane&15); lanes 0-15 hold K{0..7,16..23}, lanes 16-31 K{8..15,24..31}.
__device__ __forceinline__ BFrag load_b32x16(const bf16_t* __restrict__ baseT,
                                             int col0, int ld, int k0, int lane) {
  BFrag f;
  const int kb = k0 + ((lane & 16) ? 8 : 0);
  const bf16_t* p = baseT + (size_t)(col0 + (lane & 15)) * (size_t)ld + kb;
  f.h[0] = *(const v8bf*)(p);
  f.h[1] = *(const v8bf*)(p + 16);
  return f;
}

// ---------------- precision-conversion kernels ----------------

__global__ __launch_bounds__(256)
void cvt_f32_to_bf16(const float* __restrict__ in, bf16_t* __restrict__ out, int n) {
  int i = (blockIdx.x * blockDim.x + threadIdx.x) * 4;
  if (i + 3 < n) {
    float4 v = *(const float4*)(in + i);
    out[i + 0] = (bf16_t)v.x;
    out[i + 1] = (bf16_t)v.y;
    out[i + 2] = (bf16_t)v.z;
    out[i + 3] = (bf16_t)v.w;
  }
}

// out[n][k] = (bf16) in[k][n]  (K x N row-major -> N x K row-major)
__global__ __launch_bounds__(256)
void transpose_f32_to_bf16(const float* __restrict__ in, bf16_t* __restrict__ out,
                           int K, int N) {
  int idx = blockIdx.x * blockDim.x + threadIdx.x;
  if (idx >= K * N) return;
  int nrow = idx / K;
  int k    = idx - nrow * K;
  out[idx] = (bf16_t)in[(size_t)k * (size_t)N + nrow];
}

// ---------------- QKV projection GEMM ----------------
// C(8192x3072) = Xb(8192x1024) * Wqkv + b.
// 8 waves/block, each wave computes a 32x64 tile (2x4 register blocking,
// 8 WMMAs per K-step, ~96 B of L2 traffic per output element).
// Epilogue scatters Q,K -> (B,H,T,D) and V -> (B,H,D,T) (transposed) bf16.

__global__ __launch_bounds__(256)
void qkv_gemm_kernel(const bf16_t* __restrict__ xb, const bf16_t* __restrict__ wT,
                     const float* __restrict__ bias,
                     bf16_t* __restrict__ Q, bf16_t* __restrict__ K,
                     bf16_t* __restrict__ Vt) {
  const int lane = threadIdx.x & 31;
  const int wid  = threadIdx.x >> 5;
  const int gw   = blockIdx.x * 8 + wid;
  const int nTiles = NQKV / 64;           // 48
  const int mT = gw / nTiles, nT = gw - mT * nTiles;
  const int m0 = mT * 32, n0 = nT * 64;
  const int hi = (lane >> 4) & 1;

  v8f acc[2][4];
  #pragma unroll
  for (int a = 0; a < 2; a++)
    #pragma unroll
    for (int b = 0; b < 4; b++)
      acc[a][b] = (v8f){0.f,0.f,0.f,0.f,0.f,0.f,0.f,0.f};

  for (int k0 = 0; k0 < CDIM; k0 += 32) {
    // L2 prefetch hint for the next K-slab (global_prefetch_b8; speculative)
    __builtin_prefetch((const void*)(xb + (size_t)(m0 + (lane & 15)) * CDIM + k0 + 32), 0, 1);
    BFrag a0 = load_a16x32(xb, m0,      CDIM, k0, lane);
    BFrag a1 = load_a16x32(xb, m0 + 16, CDIM, k0, lane);
    #pragma unroll
    for (int in = 0; in < 4; in++) {
      BFrag bb = load_b32x16(wT, n0 + in * 16, CDIM, k0, lane);
      acc[0][in] = wmma_bf16(a0.v, bb.v, acc[0][in]);
      acc[1][in] = wmma_bf16(a1.v, bb.v, acc[1][in]);
    }
  }

  #pragma unroll
  for (int im = 0; im < 2; im++) {
    #pragma unroll
    for (int in = 0; in < 4; in++) {
      #pragma unroll
      for (int r = 0; r < 8; r++) {
        const int m   = m0 + im * 16 + r + hi * 8;
        const int col = n0 + in * 16 + (lane & 15);
        float v = acc[im][in][r] + bias[col];
        const int b = m >> 11, t = m & (TSEQ - 1);
        const int sect = col >> 10, c = col & (CDIM - 1);
        const int h = c >> 6, d = c & (HDIM - 1);
        const size_t bh = (size_t)(b * NHEAD + h);
        if (sect == 0) {
          Q[(bh * TSEQ + t) * HDIM + d] = (bf16_t)v;
        } else if (sect == 1) {
          K[(bh * TSEQ + t) * HDIM + d] = (bf16_t)v;
        } else {
          Vt[(bh * HDIM + d) * TSEQ + t] = (bf16_t)v;   // transposed store
        }
      }
    }
  }
}

// ---------------- flash attention (causal, online softmax) ----------------
// One wave per 16-query tile; 8 waves (128 queries) per block; 16 blocks per
// (b,h). Score and PV matmuls are WMMA; P transposed C->A layout via LDS.
// V is consumed from its (B,H,D,T) transposed layout so the PV B-fragment is
// two contiguous b128 loads per lane (keys contiguous down each d-column).

__global__ __launch_bounds__(256)
void attn_kernel(const bf16_t* __restrict__ Q, const bf16_t* __restrict__ K,
                 const bf16_t* __restrict__ Vt, bf16_t* __restrict__ O) {
  __shared__ bf16_t lds_p[8][16 * 32];   // 1KB per wave P-transpose scratch

  const int lane = threadIdx.x & 31;
  const int wid  = threadIdx.x >> 5;
  const int hi   = (lane >> 4) & 1;
  const int bh   = blockIdx.x >> 4;          // 0..63
  const int qblk = blockIdx.x & 15;
  const int q0   = qblk * 128 + wid * 16;

  const bf16_t* Qp = Q  + (size_t)bh * TSEQ * HDIM;
  const bf16_t* Kp = K  + (size_t)bh * TSEQ * HDIM;
  const bf16_t* Vp = Vt + (size_t)bh * HDIM * TSEQ;  // (D,T)

  // Q fragments for the two 32-wide reduction chunks of D=64
  BFrag qa[2];
  qa[0] = load_a16x32(Qp, q0, HDIM, 0,  lane);
  qa[1] = load_a16x32(Qp, q0, HDIM, 32, lane);

  v8f o[4];
  #pragma unroll
  for (int dt = 0; dt < 4; dt++) o[dt] = (v8f){0.f,0.f,0.f,0.f,0.f,0.f,0.f,0.f};
  float mrun[8], lrun[8];
  #pragma unroll
  for (int r = 0; r < 8; r++) { mrun[r] = -1e30f; lrun[r] = 0.f; }

  for (int kc = 0; kc < q0 + 16; kc += 32) {
    // ---- scores S (16x32) = Q(16x64) * K^T  : 4 WMMAs ----
    v8f s[2];
    #pragma unroll
    for (int kt = 0; kt < 2; kt++) {
      BFrag kb0 = load_b32x16(Kp, kc + kt * 16, HDIM, 0,  lane);
      BFrag kb1 = load_b32x16(Kp, kc + kt * 16, HDIM, 32, lane);
      v8f t = (v8f){0.f,0.f,0.f,0.f,0.f,0.f,0.f,0.f};
      t = wmma_bf16(qa[0].v, kb0.v, t);
      t = wmma_bf16(qa[1].v, kb1.v, t);
      s[kt] = t;
    }

    // ---- scale + causal mask ----
    #pragma unroll
    for (int r = 0; r < 8; r++) {
      const int q = q0 + r + hi * 8;
      #pragma unroll
      for (int kt = 0; kt < 2; kt++) {
        const int key = kc + kt * 16 + (lane & 15);
        float v = s[kt][r] * 0.125f;           // 1/sqrt(64)
        s[kt][r] = (key > q) ? -1e30f : v;
      }
    }

    // ---- online softmax update (rows striped across 16 lanes) ----
    #pragma unroll
    for (int r = 0; r < 8; r++) {
      float vmax = fmaxf(s[0][r], s[1][r]);
      #pragma unroll
      for (int mm = 8; mm >= 1; mm >>= 1)
        vmax = fmaxf(vmax, __shfl_xor(vmax, mm, 16));
      const float newmax = fmaxf(mrun[r], vmax);
      const float alpha  = __expf(mrun[r] - newmax);
      const float p0 = __expf(s[0][r] - newmax);
      const float p1 = __expf(s[1][r] - newmax);
      s[0][r] = p0; s[1][r] = p1;
      float ls = p0 + p1;
      #pragma unroll
      for (int mm = 8; mm >= 1; mm >>= 1)
        ls += __shfl_xor(ls, mm, 16);
      lrun[r] = lrun[r] * alpha + ls;
      mrun[r] = newmax;
      #pragma unroll
      for (int dt = 0; dt < 4; dt++) o[dt][r] *= alpha;
    }

    // ---- transpose P from C-layout to A-layout through wave-private LDS ----
    #pragma unroll
    for (int r = 0; r < 8; r++) {
      const int m = r + hi * 8;
      lds_p[wid][m * 32 +      (lane & 15)] = (bf16_t)s[0][r];
      lds_p[wid][m * 32 + 16 + (lane & 15)] = (bf16_t)s[1][r];
    }
    BFrag pf;
    {
      const int kb = hi ? 8 : 0;
      const bf16_t* pp = &lds_p[wid][(lane & 15) * 32 + kb];
      pf.h[0] = *(const v8bf*)(pp);
      pf.h[1] = *(const v8bf*)(pp + 16);
    }

    // ---- O(16x64) += P(16x32) * V(32x64) : 4 WMMAs ----
    // V^T rows are d-columns; keys contiguous -> b128 fragment loads.
    #pragma unroll
    for (int dt = 0; dt < 4; dt++) {
      BFrag vf = load_b32x16(Vp, dt * 16, TSEQ, kc, lane);
      o[dt] = wmma_bf16(pf.v, vf.v, o[dt]);
    }
  }

  // ---- epilogue: normalize and store to (B,T,C) bf16 ----
  const int b = bh >> 4, h = bh & (NHEAD - 1);
  #pragma unroll
  for (int r = 0; r < 8; r++) {
    const float inv = 1.0f / lrun[r];
    const int t = q0 + r + hi * 8;
    #pragma unroll
    for (int dt = 0; dt < 4; dt++) {
      const int col = h * HDIM + dt * 16 + (lane & 15);
      O[((size_t)(b * TSEQ + t)) * CDIM + col] = (bf16_t)(o[dt][r] * inv);
    }
  }
}

// ---------------- output projection GEMM ----------------
// out(8192x1024 fp32) = attnO(bf16) * Wout + b_out; 32x64 tile per wave.

__global__ __launch_bounds__(256)
void out_gemm_kernel(const bf16_t* __restrict__ ab, const bf16_t* __restrict__ wT,
                     const float* __restrict__ bias, float* __restrict__ out) {
  const int lane = threadIdx.x & 31;
  const int wid  = threadIdx.x >> 5;
  const int gw   = blockIdx.x * 8 + wid;
  const int nTiles = CDIM / 64;           // 16
  const int mT = gw / nTiles, nT = gw - mT * nTiles;
  const int m0 = mT * 32, n0 = nT * 64;
  const int hi = (lane >> 4) & 1;

  v8f acc[2][4];
  #pragma unroll
  for (int a = 0; a < 2; a++)
    #pragma unroll
    for (int b = 0; b < 4; b++)
      acc[a][b] = (v8f){0.f,0.f,0.f,0.f,0.f,0.f,0.f,0.f};

  for (int k0 = 0; k0 < CDIM; k0 += 32) {
    __builtin_prefetch((const void*)(ab + (size_t)(m0 + (lane & 15)) * CDIM + k0 + 32), 0, 1);
    BFrag a0 = load_a16x32(ab, m0,      CDIM, k0, lane);
    BFrag a1 = load_a16x32(ab, m0 + 16, CDIM, k0, lane);
    #pragma unroll
    for (int in = 0; in < 4; in++) {
      BFrag bb = load_b32x16(wT, n0 + in * 16, CDIM, k0, lane);
      acc[0][in] = wmma_bf16(a0.v, bb.v, acc[0][in]);
      acc[1][in] = wmma_bf16(a1.v, bb.v, acc[1][in]);
    }
  }

  #pragma unroll
  for (int im = 0; im < 2; im++) {
    #pragma unroll
    for (int in = 0; in < 4; in++) {
      #pragma unroll
      for (int r = 0; r < 8; r++) {
        const int m   = m0 + im * 16 + r + hi * 8;
        const int col = n0 + in * 16 + (lane & 15);
        out[(size_t)m * CDIM + col] = acc[im][in][r] + bias[col];
      }
    }
  }
}

// ---------------- host launcher ----------------

extern "C" void kernel_launch(void* const* d_in, const int* in_sizes, int n_in,
                              void* d_out, int out_size, void* d_ws, size_t ws_size,
                              hipStream_t stream) {
  const float* x     = (const float*)d_in[0];
  const float* w_qkv = (const float*)d_in[1];
  const float* b_qkv = (const float*)d_in[2];
  const float* w_out = (const float*)d_in[3];
  const float* b_out = (const float*)d_in[4];
  float* out = (float*)d_out;

  // workspace carve-up (bf16 elements), ~92 MB total
  bf16_t* p = (bf16_t*)d_ws;
  bf16_t* xb    = p; p += (size_t)MROWS * CDIM;   // x in bf16
  bf16_t* wqkvT = p; p += (size_t)NQKV  * CDIM;   // w_qkv^T bf16 (N-major)
  bf16_t* woutT = p; p += (size_t)CDIM  * CDIM;   // w_out^T bf16
  bf16_t* Qh    = p; p += (size_t)MROWS * CDIM;   // (B,H,T,D)
  bf16_t* Kh    = p; p += (size_t)MROWS * CDIM;   // (B,H,T,D)
  bf16_t* Vth   = p; p += (size_t)MROWS * CDIM;   // (B,H,D,T)  transposed V
  bf16_t* attnO = p; p += (size_t)MROWS * CDIM;   // (B,T,C)

  {
    const int n = MROWS * CDIM;                   // 8,388,608
    cvt_f32_to_bf16<<<(n / 4 + 255) / 256, 256, 0, stream>>>(x, xb, n);
  }
  {
    const int n = NQKV * CDIM;                    // 3,145,728
    transpose_f32_to_bf16<<<(n + 255) / 256, 256, 0, stream>>>(w_qkv, wqkvT, CDIM, NQKV);
  }
  {
    const int n = CDIM * CDIM;                    // 1,048,576
    transpose_f32_to_bf16<<<(n + 255) / 256, 256, 0, stream>>>(w_out, woutT, CDIM, CDIM);
  }

  // (8192/32)*(3072/64) = 12288 wave-tiles / 8 waves per block
  qkv_gemm_kernel<<<1536, 256, 0, stream>>>(xb, wqkvT, b_qkv, Qh, Kh, Vth);

  // 64 (b,h) * 16 query-blocks
  attn_kernel<<<1024, 256, 0, stream>>>(Qh, Kh, Vth, attnO);

  // (8192/32)*(1024/64) = 4096 wave-tiles / 8 waves per block
  out_gemm_kernel<<<512, 256, 0, stream>>>(attnO, woutT, b_out, out);
}
